// MultiHeadAttention_65163243815039
// MI455X (gfx1250) — matrix operand order repaired and checked by reading, not compiled, not run on hardware
//
#include <hip/hip_runtime.h>
#include <hip/hip_bf16.h>

// MHA: B=8, Lq=Lk=1024, D=1024, H=8, dh=128
#define BATCH 8
#define LSEQ  1024
#define DMODEL 1024
#define NHEAD 8
#define DHEAD 128

typedef __attribute__((ext_vector_type(16))) __bf16    v16bf;
typedef __attribute__((ext_vector_type(8)))  float     v8f;
typedef __attribute__((ext_vector_type(8)))  unsigned  v8u;

__device__ __forceinline__ v16bf as_v16bf(v8u u) {
  union { v8u u; v16bf b; } o; o.u = u; return o.b;
}

// ---- f32 pair -> packed bf16x2 ----
#if __has_builtin(__builtin_amdgcn_cvt_pk_bf16_f32)
typedef __attribute__((ext_vector_type(2))) __bf16 v2bf;
__device__ __forceinline__ unsigned pack2bf(float lo, float hi) {
  v2bf t = __builtin_amdgcn_cvt_pk_bf16_f32(lo, hi);
  union { v2bf b; unsigned u; } o; o.b = t; return o.u;
}
#elif __has_builtin(__builtin_amdgcn_perm)
__device__ __forceinline__ unsigned pack2bf(float lo, float hi) {
  union { float f; unsigned u; } a, b; a.f = lo; b.f = hi;
  // round-half-up then take the two high bytes of each: 3 VALU ops total
  return __builtin_amdgcn_perm(b.u + 0x8000u, a.u + 0x8000u, 0x07060302u);
}
#else
__device__ __forceinline__ unsigned rnd_bf16(float f) {
  union { float f; unsigned u; } v; v.f = f;
  return (v.u + 0x7FFFu + ((v.u >> 16) & 1u)) >> 16;
}
__device__ __forceinline__ unsigned pack2bf(float lo, float hi) {
  return rnd_bf16(lo) | (rnd_bf16(hi) << 16);
}
#endif

// =========== WMMA fragment loaders (ISA 7.12.2), dword-packed ===========
// A fragment (16x32 MxK) from bf16 (ushort) row-major source.
__device__ __forceinline__ v16bf load_a_bf(const unsigned short* X, int ld, int m0, int k0) {
  int lane = threadIdx.x & 31, half = lane >> 4;
  const unsigned short* p = X + (size_t)(m0 + (lane & 15)) * ld + k0;
  v8u a;
#pragma unroll
  for (int v = 0; v < 8; ++v) {
    int k = (v < 4) ? (half * 8 + v * 2) : (16 + half * 8 + (v - 4) * 2);
    a[v] = *(const unsigned*)(p + k);
  }
  return as_v16bf(a);
}
// B fragment (32x16 KxN), B[k,n] = X[n0+n][k0+k], bf16 source (K/Vt rows).
__device__ __forceinline__ v16bf load_b_t_bf(const unsigned short* X, int ld, int n0, int k0) {
  int lane = threadIdx.x & 31;
  const unsigned short* p = X + (size_t)(n0 + (lane & 15)) * ld + k0 + (lane >> 4) * 8;
  v8u b;
#pragma unroll
  for (int v = 0; v < 8; ++v) b[v] = *(const unsigned*)(p + ((v < 4) ? 2 * v : 16 + 2 * (v - 4)));
  return as_v16bf(b);
}
// A fragment from f32 row-major source (LDS panels), pair-packed.
__device__ __forceinline__ v16bf load_a_f32(const float* X, int ld, int m0, int k0) {
  int lane = threadIdx.x & 31, half = lane >> 4;
  const float* p = X + (size_t)(m0 + (lane & 15)) * ld + k0;
  v8u a;
#pragma unroll
  for (int v = 0; v < 8; ++v) {
    int k = (v < 4) ? (half * 8 + v * 2) : (16 + half * 8 + (v - 4) * 2);
    a[v] = pack2bf(p[k], p[k + 1]);
  }
  return as_v16bf(a);
}
// B fragment (transposed source) from f32 (LDS panels).
__device__ __forceinline__ v16bf load_b_t_f32(const float* X, int ld, int n0, int k0) {
  int lane = threadIdx.x & 31;
  const float* p = X + (size_t)(n0 + (lane & 15)) * ld + k0 + (lane >> 4) * 8;
  v8u b;
#pragma unroll
  for (int v = 0; v < 8; ++v) { int k = (v < 4) ? 2 * v : 16 + 2 * (v - 4); b[v] = pack2bf(p[k], p[k + 1]); }
  return as_v16bf(b);
}
__device__ __forceinline__ v8f wmma_bf16(v16bf a, v16bf b, v8f c) {
  return __builtin_amdgcn_wmma_f32_16x16x32_bf16(false, a, false, b, (short)0, c, false, false);
}

// ---- CDNA5 async global->LDS copy (ASYNCcnt) ----
__device__ __forceinline__ void async_cp16(unsigned lds_off, const float* gsrc) {
  asm volatile("global_load_async_to_lds_b128 %0, %1, off"
               :: "v"(lds_off), "v"(gsrc) : "memory");
}
__device__ __forceinline__ void async_wait0() {
  asm volatile("s_wait_asynccnt 0" ::: "memory");
}

// =====================================================================
// Kernel 1: Q/K/V projections  y = x @ W.T  (unchanged structure;
// double-buffered async-to-LDS staging; V stored transposed).
// =====================================================================
#define APAD 36
#define A_FLTS (128 * APAD)
#define B_FLTS (64 * APAD)
#define QKV_LDS ((2 * A_FLTS + 2 * B_FLTS) * 4)  // 55296 bytes

__device__ __forceinline__ void stage_tiles(const float* X, const float* W,
                                            int m0, int n0, int kb, int buf) {
  int t = threadIdx.x;
  unsigned aoff = (unsigned)buf * (A_FLTS * 4);
  unsigned boff = (unsigned)(2 * A_FLTS * 4) + (unsigned)buf * (B_FLTS * 4);
#pragma unroll
  for (int j = 0; j < 4; ++j) {
    int idx = j * 256 + t;
    int row = idx >> 3, c4 = (idx & 7) * 4;
    async_cp16(aoff + (unsigned)(row * APAD + c4) * 4u,
               X + (size_t)(m0 + row) * DMODEL + kb + c4);
  }
#pragma unroll
  for (int j = 0; j < 2; ++j) {
    int idx = j * 256 + t;
    int row = idx >> 3, c4 = (idx & 7) * 4;
    async_cp16(boff + (unsigned)(row * APAD + c4) * 4u,
               W + (size_t)(n0 + row) * DMODEL + kb + c4);
  }
}

__global__ void qkv_proj_kernel(const float* __restrict__ query,
                                const float* __restrict__ keys,
                                const float* __restrict__ Wq,
                                const float* __restrict__ Wk,
                                const float* __restrict__ Wv,
                                unsigned short* __restrict__ Qb,
                                unsigned short* __restrict__ Kb,
                                unsigned short* __restrict__ Vt) {
  extern __shared__ float smem[];

  int which = blockIdx.z;
  const float* X = (which == 0) ? query : keys;
  const float* W = (which == 0) ? Wq : (which == 1) ? Wk : Wv;

  int n0 = blockIdx.x * 64;
  int m0 = blockIdx.y * 128;
  int wave = threadIdx.x >> 5, lane = threadIdx.x & 31;
  int wm = wave >> 1, wn = wave & 1;

  stage_tiles(X, W, m0, n0, 0, 0);
  v8f acc[2][2] = {};
  for (int kb = 0; kb < DMODEL; kb += 32) {
    int cur = (kb >> 5) & 1;
    async_wait0();
    __syncthreads();
    if (kb + 32 < DMODEL)
      stage_tiles(X, W, m0, n0, kb + 32, cur ^ 1);
    const float* Ac = smem + cur * A_FLTS;
    const float* Bc = smem + 2 * A_FLTS + cur * B_FLTS;
    v16bf a0 = load_a_f32(Ac, APAD, wm * 32, 0);
    v16bf a1 = load_a_f32(Ac, APAD, wm * 32 + 16, 0);
    v16bf b0 = load_b_t_f32(Bc, APAD, wn * 32, 0);
    v16bf b1 = load_b_t_f32(Bc, APAD, wn * 32 + 16, 0);
    acc[0][0] = wmma_bf16(a0, b0, acc[0][0]);
    acc[0][1] = wmma_bf16(a0, b1, acc[0][1]);
    acc[1][0] = wmma_bf16(a1, b0, acc[1][0]);
    acc[1][1] = wmma_bf16(a1, b1, acc[1][1]);
  }
  int colb = n0 + wn * 32 + (lane & 15);
  int rowb = m0 + wm * 32 + (lane >> 4) * 8;
  if (which < 2) {
    unsigned short* Dst = (which == 0) ? Qb : Kb;
#pragma unroll
    for (int i = 0; i < 2; ++i)
#pragma unroll
      for (int j = 0; j < 2; ++j)
#pragma unroll
        for (int r = 0; r < 8; ++r)
          Dst[(size_t)(rowb + i * 16 + r) * DMODEL + colb + j * 16] =
              (unsigned short)(pack2bf(acc[i][j][r], 0.f) & 0xFFFFu);
  } else {
    int bidx = m0 >> 10;
    int l0 = rowb & 1023;
#pragma unroll
    for (int i = 0; i < 2; ++i)
#pragma unroll
      for (int j = 0; j < 2; ++j) {
        unsigned short* col = Vt + (size_t)bidx * DMODEL * LSEQ +
                              (size_t)(colb + j * 16) * LSEQ + l0 + i * 16;
#pragma unroll
        for (int r = 0; r < 4; ++r)
          *(unsigned*)(col + 2 * r) = pack2bf(acc[i][j][2 * r], acc[i][j][2 * r + 1]);
      }
  }
}

// =====================================================================
// Kernel 2: fused scores + softmax + sim@V.
// LDS = 64 KB of dwords. Per row r:
//   f32 scores (phase 1/2) at dword  SW(r,i)  = r*1024 + (i ^ 4*(r&15))
//   bf16 pairs (phase 2/3) at dword  SBW(r,kd)= r*1024 + 512 + (kd ^ 4*(r&15))
// (bf16 stripe overwrites the dead upper half of the f32 row; the row is
//  register-resident when that happens.)
// =====================================================================
#define SW(r, c)   (((r) << 10) + ((c) ^ (((r) & 15) << 2)))
#define SBW(r, kd) (((r) << 10) + 512 + ((kd) ^ (((r) & 15) << 2)))

// A fragment from the packed-bf16 LDS stripe: plain dword loads.
__device__ __forceinline__ v16bf load_a_sb(const unsigned* su, int k0) {
  int lane = threadIdx.x & 31, half = lane >> 4, row = lane & 15;
  v8u a;
#pragma unroll
  for (int v = 0; v < 8; ++v) {
    int k = k0 + ((v < 4) ? (half * 8 + 2 * v) : (16 + half * 8 + 2 * (v - 4)));
    a[v] = su[SBW(row, k >> 1)];
  }
  return as_v16bf(a);
}

__global__ void attn_fused_kernel(const unsigned short* __restrict__ Qb,
                                  const unsigned short* __restrict__ Kb,
                                  const unsigned short* __restrict__ Vt,
                                  const int* __restrict__ mask,
                                  const float* __restrict__ rel,
                                  float* __restrict__ sim,
                                  float* __restrict__ out) {
  extern __shared__ unsigned su[];  // 16 x 1024 dwords
  int qt = blockIdx.x;
  int hb = blockIdx.y;              // h*B + b
  int h = hb >> 3, b = hb & 7;
  int wave = threadIdx.x >> 5, lane = threadIdx.x & 31;

  const unsigned short* Qm = Qb + (size_t)b * LSEQ * DMODEL + (size_t)h * DHEAD;
  const unsigned short* Km = Kb + (size_t)b * LSEQ * DMODEL + (size_t)h * DHEAD;
  const unsigned short* Vm = Vt + (size_t)b * DMODEL * LSEQ + (size_t)h * DHEAD * LSEQ;
  const int*   Mb = mask + (size_t)b * LSEQ * LSEQ;
  const float* Rb = rel  + (size_t)b * LSEQ * LSEQ;

  // Q fragments loaded once, reused for all 8 key tiles of this wave.
  v16bf aq[4];
#pragma unroll
  for (int kk = 0; kk < 4; ++kk) aq[kk] = load_a_bf(Qm, DMODEL, qt * 16, kk * 32);

  // Phase 1: raw scores -> LDS (f32 bits).
#pragma unroll
  for (int t = 0; t < 8; ++t) {
    int kt = wave * 8 + t;
    v8f c = {};
#pragma unroll
    for (int kk = 0; kk < 4; ++kk)
      c = wmma_bf16(aq[kk], load_b_t_bf(Km, DMODEL, kt * 16, kk * 32), c);
    int kc = kt * 16 + (lane & 15);
    int qloc = (lane >> 4) * 8;
#pragma unroll
    for (int r = 0; r < 8; ++r) su[SW(qloc + r, kc)] = __float_as_uint(c[r]);
  }
  __syncthreads();

  // Phase 2: row register-resident softmax. Wave w owns rows 2w, 2w+1.
  // Each lane holds 32 consecutive-pair elements of the row.
  const float scale = 0.03125f;  // 1/sqrt(1024)
#pragma unroll
  for (int rr = 0; rr < 2; ++rr) {
    int row = wave * 2 + rr;
    int q = qt * 16 + row;
    const float* relrow = Rb + (size_t)q * LSEQ;
    const int*   mrow   = Mb + (size_t)q * LSEQ;
    float vals[32];
    float m = -3.4e38f;
#pragma unroll
    for (int j = 0; j < 16; ++j) {
      int i = 2 * lane + 64 * j;
      float v0 = (__uint_as_float(su[SW(row, i)])     + relrow[i])     * scale;
      float v1 = (__uint_as_float(su[SW(row, i + 1)]) + relrow[i + 1]) * scale;
      if (mrow[i]     <= 0) v0 = -1e9f;
      if (mrow[i + 1] <= 0) v1 = -1e9f;
      vals[2 * j] = v0; vals[2 * j + 1] = v1;
      m = fmaxf(m, fmaxf(v0, v1));
    }
#pragma unroll
    for (int off = 16; off > 0; off >>= 1) m = fmaxf(m, __shfl_xor(m, off, 32));
    float sum = 0.f;
#pragma unroll
    for (int j = 0; j < 32; ++j) { vals[j] = __expf(vals[j] - m); sum += vals[j]; }
#pragma unroll
    for (int off = 16; off > 0; off >>= 1) sum += __shfl_xor(sum, off, 32);
    float inv = 1.0f / sum;
    float* Sg = sim + (size_t)hb * LSEQ * LSEQ + (size_t)q * LSEQ;
#pragma unroll
    for (int j = 0; j < 16; ++j) {
      int i = 2 * lane + 64 * j;
      float v0 = vals[2 * j] * inv, v1 = vals[2 * j + 1] * inv;
      *(float2*)(Sg + i) = make_float2(v0, v1);     // one b64 store
      su[SBW(row, i >> 1)] = pack2bf(v0, v1);       // bf16 stripe for phase 3
    }
  }
  __syncthreads();

  // Phase 3: out tile = sim_stripe @ V from bf16 LDS (no conversion VALU).
  v8f c = {};
  for (int k0 = 0; k0 < LSEQ; k0 += 32) {
    v16bf a  = load_a_sb(su, k0);
    v16bf bb = load_b_t_bf(Vm, LSEQ, wave * 16, k0);
    c = wmma_bf16(a, bb, c);
  }
  int n = h * DHEAD + wave * 16 + (lane & 15);
  int qbase = qt * 16 + (lane >> 4) * 8;
#pragma unroll
  for (int r = 0; r < 8; ++r)
    out[(size_t)b * LSEQ * DMODEL + (size_t)(qbase + r) * DMODEL + n] = c[r];
}

extern "C" void kernel_launch(void* const* d_in, const int* in_sizes, int n_in,
                              void* d_out, int out_size, void* d_ws, size_t ws_size,
                              hipStream_t stream) {
  const float* query = (const float*)d_in[0];
  const float* keys  = (const float*)d_in[1];
  const int*   mask  = (const int*)d_in[2];
  const float* rel   = (const float*)d_in[3];
  const float* Wq    = (const float*)d_in[4];
  const float* Wk    = (const float*)d_in[5];
  const float* Wv    = (const float*)d_in[6];

  float* out = (float*)d_out;                       // [B, L, D]
  float* sim = out + (size_t)BATCH * LSEQ * DMODEL; // [H*B, L, L]

  unsigned short* Qb = (unsigned short*)d_ws;       // bf16, 16 MB each
  unsigned short* Kb = Qb + (size_t)BATCH * LSEQ * DMODEL;
  unsigned short* Vt = Kb + (size_t)BATCH * LSEQ * DMODEL;  // [B][D][L]

  qkv_proj_kernel<<<dim3(DMODEL / 64, (BATCH * LSEQ) / 128, 3), 256, QKV_LDS, stream>>>(
      query, keys, Wq, Wk, Wv, Qb, Kb, Vt);
  attn_fused_kernel<<<dim3(LSEQ / 16, NHEAD * BATCH), 256, 16 * LSEQ * 4, stream>>>(
      Qb, Kb, Vt, mask, rel, sim, out);
}
